// MemoryModule_41901700939802
// MI455X (gfx1250) — matrix operand order repaired and checked by reading, not compiled, not run on hardware
//
#include <hip/hip_runtime.h>

#define NROW  65536
#define MSIZE 2048
#define MDIM  256
#define PAD   8   // 16B of f16 padding per LDS row -> rotates banks per row

typedef _Float16 v8h  __attribute__((ext_vector_type(8)));
typedef _Float16 v16h __attribute__((ext_vector_type(16)));
typedef float    v8f  __attribute__((ext_vector_type(8)));

static __device__ __forceinline__ v16h cat16(v8h lo, v8h hi) {
    return __builtin_shufflevector(lo, hi, 0,1,2,3,4,5,6,7,8,9,10,11,12,13,14,15);
}

// Prologue: memory (f32) -> row-major f16 copy and transposed f16 copy in workspace.
__global__ void memmod_cvt(const float* __restrict__ mem,
                           _Float16* __restrict__ memF16,
                           _Float16* __restrict__ memT16)
{
    int idx = blockIdx.x * 256 + threadIdx.x;   // 0 .. MSIZE*MDIM-1
    int j = idx >> 8;       // memory row
    int d = idx & 255;      // memory col
    float v = mem[idx];
    memF16[idx]           = (_Float16)v;
    memT16[d * MSIZE + j] = (_Float16)v;
}

// Fused: scores GEMM (f16 WMMA, f32 acc) + softmax + output GEMM.
// Block = 16 rows of x, 256 threads = 8 waves (wave32).
__global__ __launch_bounds__(256)
void memmod_fused(const float* __restrict__ x,
                  const _Float16* __restrict__ memF16,
                  const _Float16* __restrict__ memT16,
                  float* __restrict__ outVec,
                  float* __restrict__ attn)
{
    __shared__ __align__(32) _Float16 xbuf[16][MDIM + PAD];
    __shared__ __align__(32) _Float16 wbuf[16][MDIM + PAD];
    __shared__ float part[8][16];
    __shared__ float rowInv[16];

    const int t    = threadIdx.x;
    const int lane = t & 31;
    const int wave = t >> 5;
    const int half = (lane >> 4) & 1;   // 0: lanes 0-15, 1: lanes 16-31
    const int l16  = lane & 15;
    const int i0   = blockIdx.x * 16;

    const int aBase = half * 16;        // A-operand K offset (sequential pattern)
    const int bBase = half * 16;        // B-operand K offset (sequential pattern)

    // ---- stage x tile, f32 -> f16 ----
    #pragma unroll
    for (int i = 0; i < 16; ++i)
        xbuf[i][t] = (_Float16)x[(size_t)(i0 + i) * MDIM + t];
    __syncthreads();

    // ---- phase 1: raw scores -> attn region, per-row sum(exp) ----
    float sums[8];
    #pragma unroll
    for (int v = 0; v < 8; ++v) sums[v] = 0.f;

    for (int ct = 0; ct < 16; ++ct) {
        const int j0 = wave * 256 + ct * 16;   // this wave's score column tile
        v8f c = {};
        #pragma unroll
        for (int kk = 0; kk < 8; ++kk) {
            const int k0 = kk * 32;
            v8h a0 = *(const v8h*)&xbuf[l16][k0 + aBase];
            v8h a1 = *(const v8h*)&xbuf[l16][k0 + aBase + 8];
            v16h A = cat16(a0, a1);
            v16h B = *(const v16h*)&memF16[(size_t)(j0 + l16) * MDIM + k0 + bBase];
            c = __builtin_amdgcn_wmma_f32_16x16x32_f16(false, A, false, B,
                                                       (short)0, c, false, false);
        }
        #pragma unroll
        for (int v = 0; v < 8; ++v) {
            const int row = v + half * 8;
            attn[(size_t)(i0 + row) * MSIZE + j0 + l16] = c[v];  // raw score
            sums[v] += __expf(c[v]);
        }
    }

    // reduce sum(exp) across the 16 lanes sharing the same rows
    #pragma unroll
    for (int m = 1; m <= 8; m <<= 1) {
        #pragma unroll
        for (int v = 0; v < 8; ++v)
            sums[v] += __shfl_xor(sums[v], m, 32);
    }
    if (l16 == 0) {
        #pragma unroll
        for (int v = 0; v < 8; ++v)
            part[wave][v + half * 8] = sums[v];
    }
    __threadfence();     // make raw-score stores visible before phase-2 re-read
    __syncthreads();

    if (t < 16) {
        float s = 0.f;
        #pragma unroll
        for (int w = 0; w < 8; ++w) s += part[w][t];
        rowInv[t] = 1.0f / s;
    }
    __syncthreads();

    // ---- phase 2: normalize weights + out = W @ mem ----
    v8f acc0 = {};
    v8f acc1 = {};
    const int d0a = wave * 16;          // this wave's two output col-tiles
    const int d0b = wave * 16 + 128;

    for (int kc = 0; kc < 8; ++kc) {    // 256-wide chunks over the 2048 weights
        #pragma unroll
        for (int i = 0; i < 16; ++i) {
            const size_t gi = (size_t)(i0 + i) * MSIZE + kc * 256 + t;
            float s = attn[gi];                  // raw score (L2-hot)
            float w = __expf(s) * rowInv[i];
            attn[gi]   = w;                      // final attention weight
            wbuf[i][t] = (_Float16)w;            // f16 operand for GEMM2
        }
        __syncthreads();
        #pragma unroll
        for (int kk = 0; kk < 8; ++kk) {
            const int k0 = kk * 32;
            v8h a0 = *(const v8h*)&wbuf[l16][k0 + aBase];
            v8h a1 = *(const v8h*)&wbuf[l16][k0 + aBase + 8];
            v16h A  = cat16(a0, a1);
            const int jg = kc * 256 + k0 + bBase;
            v16h B0 = *(const v16h*)&memT16[(size_t)(d0a + l16) * MSIZE + jg];
            v16h B1 = *(const v16h*)&memT16[(size_t)(d0b + l16) * MSIZE + jg];
            acc0 = __builtin_amdgcn_wmma_f32_16x16x32_f16(false, A, false, B0,
                                                          (short)0, acc0, false, false);
            acc1 = __builtin_amdgcn_wmma_f32_16x16x32_f16(false, A, false, B1,
                                                          (short)0, acc1, false, false);
        }
        __syncthreads();
    }

    #pragma unroll
    for (int v = 0; v < 8; ++v) {
        const int row = v + half * 8;
        outVec[(size_t)(i0 + row) * MDIM + d0a + l16] = acc0[v];
        outVec[(size_t)(i0 + row) * MDIM + d0b + l16] = acc1[v];
    }
}

extern "C" void kernel_launch(void* const* d_in, const int* in_sizes, int n_in,
                              void* d_out, int out_size, void* d_ws, size_t ws_size,
                              hipStream_t stream)
{
    const float* x   = (const float*)d_in[0];   // [65536, 256]
    const float* mem = (const float*)d_in[1];   // [2048, 256]

    float* outVec = (float*)d_out;                               // [65536,256]
    float* attn   = (float*)d_out + (size_t)NROW * MDIM;         // [65536,2048]

    _Float16* memF16 = (_Float16*)d_ws;                          // 1 MB
    _Float16* memT16 = (_Float16*)d_ws + (size_t)MSIZE * MDIM;   // 1 MB

    memmod_cvt<<<(MSIZE * MDIM) / 256, 256, 0, stream>>>(mem, memF16, memT16);
    memmod_fused<<<NROW / 16, 256, 0, stream>>>(x, memF16, memT16, outVec, attn);
}
